// ODE_45990509805580
// MI455X (gfx1250) — matrix operand order repaired and checked by reading, not compiled, not run on hardware
//
#include <hip/hip_runtime.h>
#include <math.h>

// ---------------------------------------------------------------------------
// Hypernetwork MLP for MI455X (gfx1250, wave32, WMMA f16->f32)
//
//   params = controls @ hyper_W^T + hyper_b        (never materialized!)
//   h   = tanh(W1 z + b1);  out = tanh(W2 h + b2)
//
// Two bilinear GEMMs with on-the-fly A fragments (A = splat * ctrl fragment):
//   stage1: K = 64*32 (+64 z-bias +32 ctrl-bias +const), N = 256
//   stage2: K = 256*32 (+256 h-bias +32 ctrl-bias +const), N = 64
// B matrices repacked once into fragment-ready f16 in d_ws (~2.2 MB, shared
// by all 512 tiles -> L2 resident).
//
// Parallelism: 512 workgroups x 4 waves. Stage 1 is N-split across the 4
// waves (acc = 4 N-tiles each); stage 2 is K-split with an LDS ds_add_f32
// reduction. 2048 waves total, ~100 VGPRs/wave.
// ---------------------------------------------------------------------------

typedef __attribute__((ext_vector_type(16))) _Float16 v16h;
typedef __attribute__((ext_vector_type(8)))  float    v8f;
typedef __attribute__((ext_vector_type(2)))  _Float16 h2;
typedef __attribute__((ext_vector_type(4)))  float    f4;

#define LATENT  64
#define HIDDEN  256
#define CONTROL 32

#define K1_CHUNKS 67    // 64 main (z*ctrl) + 2 (z bias) + 1 (ctrl bias)
#define K2_CHUNKS 265   // 256 main (h*ctrl) + 8 (h bias) + 1 (ctrl bias)
#define NT1 16          // HIDDEN / 16
#define NT2 4           // LATENT / 16

#define B1_HALFS (K1_CHUNKS * NT1 * 32 * 16)   // 548864 f16
#define B2_HALFS (K2_CHUNKS * NT2 * 32 * 16)   // 542720 f16

union AFrag { v16h v; h2 p[8]; _Float16 e[16]; };

// ---------------------------------------------------------------------------
// Pack hyper_W / hyper_b into fragment-ready f16 B matrices.
// B-fragment layout (ISA 7.12.2, 16-bit B 32x16): lane L holds N = L&15,
// half t holds K = (t&7) + 8*(L>>4) + 16*(t>>3).  Fragment (chunk j, n-tile nt) stored as
// 32 lanes x 16 halfs contiguous => one aligned 32B load per lane.
// ---------------------------------------------------------------------------
__global__ void hyper_pack_B1(const float* __restrict__ hw,
                              const float* __restrict__ hb,
                              _Float16* __restrict__ o) {
  int tid = blockIdx.x * blockDim.x + threadIdx.x;
  if (tid >= B1_HALFS) return;
  int t    = tid & 15;
  int lane = (tid >> 4) & 31;
  int nt   = (tid >> 9) & 15;
  int j    = tid >> 13;                        // 16*32*16 = 8192 = 2^13
  int n = nt * 16 + (lane & 15);               // hidden index h
  int k = 32 * j + (t & 7) + 8 * (lane >> 4) + 16 * (t >> 3);       // global K
  float v;
  if (k < 2048)        v = hw[(size_t)((n << 6) + (k >> 5)) * CONTROL + (k & 31)]; // W1
  else if (k < 2112)   v = hb[(n << 6) + (k - 2048)];                              // hyper_b W1 (dot z)
  else                 v = hw[(size_t)(16384 + n) * CONTROL + (k - 2112)];         // b1 (dot controls)
  o[tid] = (_Float16)v;
}

__global__ void hyper_pack_B2(const float* __restrict__ hw,
                              const float* __restrict__ hb,
                              _Float16* __restrict__ o) {
  int tid = blockIdx.x * blockDim.x + threadIdx.x;
  if (tid >= B2_HALFS) return;
  int t    = tid & 15;
  int lane = (tid >> 4) & 31;
  int nt   = (tid >> 9) & 3;
  int j    = tid >> 11;                        // 4*32*16 = 2048 = 2^11
  int n = nt * 16 + (lane & 15);               // latent index l
  int k = 32 * j + (t & 7) + 8 * (lane >> 4) + 16 * (t >> 3);
  float v;
  if (k < 8192)        v = hw[(size_t)(16640 + (n << 8) + (k >> 5)) * CONTROL + (k & 31)]; // W2
  else if (k < 8448)   v = hb[16640 + (n << 8) + (k - 8192)];                              // hyper_b W2 (dot h)
  else                 v = hw[(size_t)(33024 + n) * CONTROL + (k - 8448)];                 // b2 (dot controls)
  o[tid] = (_Float16)v;
}

// ---------------------------------------------------------------------------
// Main fused kernel: 1 workgroup (4 waves) == one 16-row batch tile.
// ---------------------------------------------------------------------------
#define WPT 4                   // waves per tile

__global__ __launch_bounds__(WPT * 32) void hyper_mlp(
    const float* __restrict__ controls, const float* __restrict__ z,
    const float* __restrict__ hyper_b,
    const _Float16* __restrict__ B1h, const _Float16* __restrict__ B2h,
    float* __restrict__ out) {
  __shared__ float z_s[16 * LATENT];           //  4 KB
  __shared__ float h1_s[16 * HIDDEN];          // 16 KB
  __shared__ float out_s[16 * LATENT];         //  4 KB

  const int tid  = threadIdx.x;
  const int lane = tid & 31;
  const int wv   = tid >> 5;
  const int row0 = blockIdx.x * 16;
  const int m    = lane & 15;                  // N-lane / batch row within tile
  const int g    = lane >> 4;                  // lane half-group

  // ---- controls fragment: A-layout of 16x32 tile, K(t) = (t&7) + 8g + 16*(t>>3)
  const float* crow = controls + (size_t)(row0 + m) * CONTROL;
  f4 c0 = *(const f4*)(crow + 8 * g);
  f4 c1 = *(const f4*)(crow + 8 * g + 4);
  f4 c2 = *(const f4*)(crow + 8 * g + 16);
  f4 c3 = *(const f4*)(crow + 8 * g + 20);
  AFrag ctrl;
#pragma unroll
  for (int i = 0; i < 4; ++i) {
    ctrl.e[i]      = (_Float16)c0[i];
    ctrl.e[4 + i]  = (_Float16)c1[i];
    ctrl.e[8 + i]  = (_Float16)c2[i];
    ctrl.e[12 + i] = (_Float16)c3[i];
  }

  // ---- z A-fragments for the two z-bias chunks (K = 2048..2111)
  const float* zrow = z + (size_t)(row0 + m) * LATENT;
  AFrag zf0, zf1;
  {
    f4 a0 = *(const f4*)(zrow + 8 * g);
    f4 a1 = *(const f4*)(zrow + 8 * g + 4);
    f4 a2 = *(const f4*)(zrow + 8 * g + 16);
    f4 a3 = *(const f4*)(zrow + 8 * g + 20);
    f4 b0 = *(const f4*)(zrow + 32 + 8 * g);
    f4 b1 = *(const f4*)(zrow + 32 + 8 * g + 4);
    f4 b2 = *(const f4*)(zrow + 32 + 8 * g + 16);
    f4 b3 = *(const f4*)(zrow + 32 + 8 * g + 20);
#pragma unroll
    for (int i = 0; i < 4; ++i) {
      zf0.e[i] = (_Float16)a0[i]; zf0.e[4 + i] = (_Float16)a1[i];
      zf0.e[8 + i] = (_Float16)a2[i]; zf0.e[12 + i] = (_Float16)a3[i];
      zf1.e[i] = (_Float16)b0[i]; zf1.e[4 + i] = (_Float16)b1[i];
      zf1.e[8 + i] = (_Float16)b2[i]; zf1.e[12 + i] = (_Float16)b3[i];
    }
  }

  // ---- cooperative loads: z tile -> LDS, zero the stage-2 reduction tile
  {
    int idx = tid * 8;                         // 128 threads x 8 floats = 16x64
    int r = idx >> 6, c = idx & 63;
    *(f4*)(&z_s[idx])     = *(const f4*)(z + (size_t)(row0 + r) * LATENT + c);
    *(f4*)(&z_s[idx + 4]) = *(const f4*)(z + (size_t)(row0 + r) * LATENT + c + 4);
    f4 zz4 = {0.f, 0.f, 0.f, 0.f};
    *(f4*)(&out_s[idx])     = zz4;
    *(f4*)(&out_s[idx + 4]) = zz4;
  }
  __syncthreads();

  const v8f vzero = {0.f, 0.f, 0.f, 0.f, 0.f, 0.f, 0.f, 0.f};

  // ================= stage 1 (N-split): wave wv owns N-tiles 4wv..4wv+3 =====
  v8f acc[4];
#pragma unroll
  for (int i = 0; i < 4; ++i) acc[i] = vzero;

  const v16h* B1v = (const v16h*)B1h;
  for (int j = 0; j < 64; ++j) {
    _Float16 zh = (_Float16)z_s[m * LATENT + j];
    h2 zz = {zh, zh};
    AFrag a;
#pragma unroll
    for (int p = 0; p < 8; ++p) a.p[p] = zz * ctrl.p[p];   // v_pk_mul_f16
    const v16h* bp = B1v + ((size_t)j * NT1 + 4 * wv) * 32 + lane;
    __builtin_prefetch(bp + NT1 * 32, 0, 1);               // next chunk, same nt base
#pragma unroll
    for (int i = 0; i < 4; ++i)
      acc[i] = __builtin_amdgcn_wmma_f32_16x16x32_f16(
          false, a.v, false, bp[i * 32], (short)0, acc[i], false, false);
  }
  {  // bias tail chunks: A = z-frag0, z-frag1, ctrl-frag
    const v16h* bp = B1v + ((size_t)64 * NT1 + 4 * wv) * 32 + lane;
#pragma unroll
    for (int i = 0; i < 4; ++i)
      acc[i] = __builtin_amdgcn_wmma_f32_16x16x32_f16(
          false, zf0.v, false, bp[i * 32], (short)0, acc[i], false, false);
    bp += NT1 * 32;
#pragma unroll
    for (int i = 0; i < 4; ++i)
      acc[i] = __builtin_amdgcn_wmma_f32_16x16x32_f16(
          false, zf1.v, false, bp[i * 32], (short)0, acc[i], false, false);
    bp += NT1 * 32;
#pragma unroll
    for (int i = 0; i < 4; ++i)
      acc[i] = __builtin_amdgcn_wmma_f32_16x16x32_f16(
          false, ctrl.v, false, bp[i * 32], (short)0, acc[i], false, false);
  }
  // epilogue 1: + hyper_b const, tanh, into LDS (C layout: M = r + 8g, N = m)
#pragma unroll
  for (int i = 0; i < 4; ++i) {
    int n = (4 * wv + i) * 16 + m;
    float bias = hyper_b[16384 + n];
#pragma unroll
    for (int r = 0; r < 8; ++r)
      h1_s[(r + 8 * g) * HIDDEN + n] = tanhf(acc[i][r] + bias);
  }
  __syncthreads();

  // ================= stage 2 (K-split): wave wv owns main chunks 64wv.. =====
  v8f acc2[4];
#pragma unroll
  for (int i = 0; i < 4; ++i) acc2[i] = vzero;

  const v16h* B2v = (const v16h*)B2h;
  const int j0 = 64 * wv;
  for (int jj = 0; jj < 64; ++jj) {
    const int j = j0 + jj;
    _Float16 hh = (_Float16)h1_s[m * HIDDEN + j];
    h2 zz = {hh, hh};
    AFrag a;
#pragma unroll
    for (int p = 0; p < 8; ++p) a.p[p] = zz * ctrl.p[p];
    const v16h* bp = B2v + (size_t)j * NT2 * 32 + lane;
    __builtin_prefetch(bp + NT2 * 32, 0, 1);
#pragma unroll
    for (int i = 0; i < 4; ++i)
      acc2[i] = __builtin_amdgcn_wmma_f32_16x16x32_f16(
          false, a.v, false, bp[i * 32], (short)0, acc2[i], false, false);
  }
  // hyper_b(W2)-dot-h bias chunks: wave wv takes jj = 2wv, 2wv+1
#pragma unroll
  for (int q = 0; q < 2; ++q) {
    const int jj = 2 * wv + q;
    AFrag a;
#pragma unroll
    for (int t = 0; t < 16; ++t)
      a.e[t] = (_Float16)h1_s[m * HIDDEN + 32 * jj + ((t & 7) + 8 * g + 16 * (t >> 3))];
    const v16h* bp = B2v + (size_t)(256 + jj) * NT2 * 32 + lane;
#pragma unroll
    for (int i = 0; i < 4; ++i)
      acc2[i] = __builtin_amdgcn_wmma_f32_16x16x32_f16(
          false, a.v, false, bp[i * 32], (short)0, acc2[i], false, false);
  }
  if (wv == 0) {  // b2-dot-controls chunk (uniform branch within wave)
    const v16h* bp = B2v + (size_t)264 * NT2 * 32 + lane;
#pragma unroll
    for (int i = 0; i < 4; ++i)
      acc2[i] = __builtin_amdgcn_wmma_f32_16x16x32_f16(
          false, ctrl.v, false, bp[i * 32], (short)0, acc2[i], false, false);
  }
  // K-split reduction into LDS (ds_add_f32)
#pragma unroll
  for (int i = 0; i < 4; ++i) {
    int n = i * 16 + m;
#pragma unroll
    for (int r = 0; r < 8; ++r)
      atomicAdd(&out_s[(r + 8 * g) * LATENT + n], acc2[i][r]);
  }
  __syncthreads();

  // ---- epilogue 2: + hyper_b const, tanh, write out (coalesced f4 stores)
  {
    int idx = tid * 8;                         // 128 threads x 8 = 16x64
    int r = idx >> 6, c = idx & 63;
    f4 s0 = *(const f4*)(&out_s[idx]);
    f4 s1 = *(const f4*)(&out_s[idx + 4]);
    f4 b0 = *(const f4*)(hyper_b + 33024 + c);
    f4 b1 = *(const f4*)(hyper_b + 33024 + c + 4);
    f4 o0, o1;
#pragma unroll
    for (int q = 0; q < 4; ++q) {
      o0[q] = tanhf(s0[q] + b0[q]);
      o1[q] = tanhf(s1[q] + b1[q]);
    }
    *(f4*)(out + (size_t)(row0 + r) * LATENT + c)     = o0;
    *(f4*)(out + (size_t)(row0 + r) * LATENT + c + 4) = o1;
  }
}

// ---------------------------------------------------------------------------
extern "C" void kernel_launch(void* const* d_in, const int* in_sizes, int n_in,
                              void* d_out, int out_size, void* d_ws, size_t ws_size,
                              hipStream_t stream) {
  const float* controls = (const float*)d_in[0];
  const float* zin      = (const float*)d_in[1];
  const float* hyper_W  = (const float*)d_in[2];
  const float* hyper_b  = (const float*)d_in[3];
  float* out = (float*)d_out;

  _Float16* B1 = (_Float16*)d_ws;           // 1,097,728 B
  _Float16* B2 = B1 + B1_HALFS;             // 1,085,440 B  (total ~2.2 MB of ws)

  hyper_pack_B1<<<(B1_HALFS + 255) / 256, 256, 0, stream>>>(hyper_W, hyper_b, B1);
  hyper_pack_B2<<<(B2_HALFS + 255) / 256, 256, 0, stream>>>(hyper_W, hyper_b, B2);

  const int nbatch = in_sizes[1] / LATENT;  // 8192
  const int tiles  = nbatch / 16;           // 512 tiles, 1 workgroup (4 waves) each
  hyper_mlp<<<tiles, WPT * 32, 0, stream>>>(controls, zin, hyper_b, B1, B2, out);
}
